// Cell_Linear_Attention_68229850465009
// MI455X (gfx1250) — matrix operand order repaired and checked by reading, not compiled, not run on hardware
//
#include <hip/hip_runtime.h>
#include <hip/hip_bf16.h>

// MI455X / gfx1250, wave32. Core matrix op: v_wmma_f32_16x16x32_bf16.
typedef __attribute__((ext_vector_type(16))) __bf16 v16bf;
typedef __attribute__((ext_vector_type(8)))  __bf16 v8bf;
typedef __attribute__((ext_vector_type(8)))  float  v8f;
typedef __attribute__((ext_vector_type(4)))  float  v4f;

#define CAT16(lo, hi) __builtin_shufflevector(lo, hi, 0,1,2,3,4,5,6,7,8,9,10,11,12,13,14,15)

// Fragment loads from K-major LDS rows (144-byte row stride, 16B aligned).
__device__ __forceinline__ v16bf ldfragA(const __bf16* row, int kb, int half)
{
    const v8bf lo = *(const v8bf*)(row + kb + half * 8);
    const v8bf hi = *(const v8bf*)(row + kb + 16 + half * 8);
    return CAT16(lo, hi);
}
__device__ __forceinline__ v16bf ldfragB(const __bf16* row, int kb, int half)
{
    const v8bf lo = *(const v8bf*)(row + kb + half * 8);
    const v8bf hi = *(const v8bf*)(row + kb + 16 + half * 8);
    return CAT16(lo, hi);
}
__device__ __forceinline__ v8bf pack8(const v4f a, const v4f b)
{
    v8bf r;
    #pragma unroll
    for (int i = 0; i < 4; ++i) { r[i] = (__bf16)a[i]; r[4 + i] = (__bf16)b[i]; }
    return r;
}

// ---------------------------------------------------------------------------
// 128x128 block tile, 512 threads = 16 waves, each wave a 2x2 quad of 16x16
// WMMA tiles (2x fragment reuse: 8 ds_load_b128 feed 4 WMMAs per kb step).
//   AT: A stored [K,M]; BT: B stored [N,K]; ABF: A bf16; OBF: C stored bf16.
// Batch offsets: z = blockIdx.z, zb = z/ZH, zh = z%ZH, off = zb*s?b + zh*s?h.
// ---------------------------------------------------------------------------
template<bool AT, bool ABF, bool BT, bool BIAS, bool OBF>
__global__ __launch_bounds__(512)
void gemm128(const void* __restrict__ Av, long lda,
             const float* __restrict__ Bf, long ldb,
             const float* __restrict__ bias,
             void* __restrict__ Cv, long ldc,
             int K, int ZH,
             long sAb, long sAh, long sBb, long sBh,
             long sCb, long sCh, long sBiasH)
{
    __shared__ __align__(16) __bf16 As[128][72];   // [m][k] K-major
    __shared__ __align__(16) __bf16 Bs[128][72];   // [n][k] K-major

    const int tid  = threadIdx.x;
    const int lane = tid & 31;
    const int wave = tid >> 5;
    const int tm   = wave >> 2;     // 4x4 grid of 32x32 wave quads
    const int tn   = wave & 3;

    const int  z  = blockIdx.z;
    const int  zb = z / ZH, zh = z % ZH;
    const long gm0 = (long)blockIdx.y * 128;
    const long gn0 = (long)blockIdx.x * 128;

    const float*  Af = (const float*)Av;
    const __bf16* Ab = (const __bf16*)Av;

    const long offA = zb * sAb + zh * sAh + (AT ? gm0 : gm0 * lda);
    const long offB = zb * sBb + zh * sBh + (BT ? gn0 * ldb : gn0);
    const long offC = zb * sCb + zh * sCh + gm0 * ldc + gn0;

    // Staging: 128x64 panel = 8192 elems, 16 per thread = 2 chunks of 8.
    const int kc = (tid & 7) * 8;    // K-contiguous chunk offset
    const int o0 = tid >> 3;         // outer row 0..63 (and +64)
    const int oc = (tid & 15) * 8;   // outer-contiguous chunk offset
    const int kk = tid >> 4;         // k row 0..31 (and +32)

    long stepA, stepB;
    const float* pA0; const float* pA1;
    if constexpr (AT) { pA0 = Af + offA + (long)kk * lda + oc;
                        pA1 = Af + offA + (long)(kk + 32) * lda + oc;
                        stepA = 64 * lda; }
    else              { pA0 = Af + offA + (long)o0 * lda + kc;
                        pA1 = Af + offA + (long)(o0 + 64) * lda + kc;
                        stepA = 64; }
    const __bf16* pA0b = Ab + (pA0 - Af);
    const __bf16* pA1b = Ab + (pA1 - Af);
    const float* pB0; const float* pB1;
    if constexpr (BT) { pB0 = Bf + offB + (long)o0 * ldb + kc;
                        pB1 = Bf + offB + (long)(o0 + 64) * ldb + kc;
                        stepB = 64; }
    else              { pB0 = Bf + offB + (long)kk * ldb + oc;
                        pB1 = Bf + offB + (long)(kk + 32) * ldb + oc;
                        stepB = 64 * ldb; }

    const int  half = lane >> 4;
    const int  l15  = lane & 15;
    const __bf16* ar0 = &As[tm * 32 + l15][0];
    const __bf16* ar1 = &As[tm * 32 + 16 + l15][0];
    const __bf16* br0 = &Bs[tn * 32 + l15][0];
    const __bf16* br1 = &Bs[tn * 32 + 16 + l15][0];

    v8f acc00 = {}, acc01 = {}, acc10 = {}, acc11 = {};

    for (int k0 = 0; k0 < K; k0 += 64) {
        // ---- issue all global loads for this panel ----
        v4f aL0 = {}, aL1 = {}, aL2 = {}, aL3 = {};
        v8bf a80 = {}, a81 = {};
        if constexpr (ABF) { a80 = *(const v8bf*)pA0b; a81 = *(const v8bf*)pA1b; }
        else { aL0 = *(const v4f*)pA0; aL1 = *(const v4f*)(pA0 + 4);
               aL2 = *(const v4f*)pA1; aL3 = *(const v4f*)(pA1 + 4); }
        const v4f bL0 = *(const v4f*)pB0, bL1 = *(const v4f*)(pB0 + 4);
        const v4f bL2 = *(const v4f*)pB1, bL3 = *(const v4f*)(pB1 + 4);
        if (k0 + 64 < K) {                             // global_prefetch_b8
            __builtin_prefetch(ABF ? (const void*)(pA0b + stepA)
                                   : (const void*)(pA0 + stepA), 0, 3);
            __builtin_prefetch(pB0 + stepB, 0, 3);
        }

        // ---- pack to bf16 and stage ----
        v8bf as0, as1;
        if constexpr (ABF) { as0 = a80; as1 = a81; }
        else { as0 = pack8(aL0, aL1); as1 = pack8(aL2, aL3); }
        const v8bf bs0 = pack8(bL0, bL1), bs1 = pack8(bL2, bL3);

        if constexpr (AT) {
            #pragma unroll
            for (int j = 0; j < 8; ++j) { As[oc + j][kk] = as0[j]; As[oc + j][kk + 32] = as1[j]; }
        } else {
            *(v8bf*)&As[o0][kc] = as0; *(v8bf*)&As[o0 + 64][kc] = as1;
        }
        if constexpr (BT) {
            *(v8bf*)&Bs[o0][kc] = bs0; *(v8bf*)&Bs[o0 + 64][kc] = bs1;
        } else {
            #pragma unroll
            for (int j = 0; j < 8; ++j) { Bs[oc + j][kk] = bs0[j]; Bs[oc + j][kk + 32] = bs1[j]; }
        }
        pA0 += stepA; pA1 += stepA; pA0b += stepA; pA1b += stepA;
        pB0 += stepB; pB1 += stepB;
        __syncthreads();

        // ---- 8 WMMAs per panel per wave, 2x fragment reuse ----
        #pragma unroll
        for (int kb = 0; kb < 64; kb += 32) {
            const v16bf a0 = ldfragA(ar0, kb, half);
            const v16bf a1 = ldfragA(ar1, kb, half);
            const v16bf b0 = ldfragB(br0, kb, half);
            const v16bf b1 = ldfragB(br1, kb, half);
            acc00 = __builtin_amdgcn_wmma_f32_16x16x32_bf16(false, a0, false, b0, (short)0, acc00, false, false);
            acc01 = __builtin_amdgcn_wmma_f32_16x16x32_bf16(false, a0, false, b1, (short)0, acc01, false, false);
            acc10 = __builtin_amdgcn_wmma_f32_16x16x32_bf16(false, a1, false, b0, (short)0, acc10, false, false);
            acc11 = __builtin_amdgcn_wmma_f32_16x16x32_bf16(false, a1, false, b1, (short)0, acc11, false, false);
        }
        __syncthreads();
    }

    // ---- store 2x2 D tiles ----
    const int col0 = tn * 32 + l15;
    const int col1 = col0 + 16;
    float b0add = 0.0f, b1add = 0.0f;
    if (BIAS) { b0add = bias[zh * sBiasH + gn0 + col0]; b1add = bias[zh * sBiasH + gn0 + col1]; }
    #pragma unroll
    for (int r = 0; r < 8; ++r) {
        const int row0 = tm * 32 + half * 8 + r;
        const int row1 = row0 + 16;
        const float v00 = acc00[r] + b0add, v01 = acc01[r] + b1add;
        const float v10 = acc10[r] + b0add, v11 = acc11[r] + b1add;
        if (OBF) {
            __bf16* C = (__bf16*)Cv;
            C[offC + (long)row0 * ldc + col0] = (__bf16)v00;
            C[offC + (long)row0 * ldc + col1] = (__bf16)v01;
            C[offC + (long)row1 * ldc + col0] = (__bf16)v10;
            C[offC + (long)row1 * ldc + col1] = (__bf16)v11;
        } else {
            float* C = (float*)Cv;
            C[offC + (long)row0 * ldc + col0] = v00;
            C[offC + (long)row0 * ldc + col1] = v01;
            C[offC + (long)row1 * ldc + col0] = v10;
            C[offC + (long)row1 * ldc + col1] = v11;
        }
    }
}

// ---------------------------------------------------------------------------
// 64x64 block tile (for the small-N batched GEMMs), 16 waves x one 16x16 tile.
// ---------------------------------------------------------------------------
template<bool AT, bool ABF, bool BT, bool BIAS, bool OBF>
__global__ __launch_bounds__(512)
void gemm64x64(const void* __restrict__ Av, long lda,
               const float* __restrict__ Bf, long ldb,
               const float* __restrict__ bias,
               void* __restrict__ Cv, long ldc,
               int K, int ZH,
               long sAb, long sAh, long sBb, long sBh,
               long sCb, long sCh, long sBiasH)
{
    __shared__ __align__(16) __bf16 As[64][72];
    __shared__ __align__(16) __bf16 Bs[64][72];

    const int tid  = threadIdx.x;
    const int lane = tid & 31;
    const int wave = tid >> 5;
    const int tm   = wave >> 2;
    const int tn   = wave & 3;

    const int  z  = blockIdx.z;
    const int  zb = z / ZH, zh = z % ZH;
    const long gm0 = (long)blockIdx.y * 64;
    const long gn0 = (long)blockIdx.x * 64;

    const float*  Af = (const float*)Av;
    const __bf16* Ab = (const __bf16*)Av;

    const long offA = zb * sAb + zh * sAh + (AT ? gm0 : gm0 * lda);
    const long offB = zb * sBb + zh * sBh + (BT ? gn0 * ldb : gn0);
    const long offC = zb * sCb + zh * sCh + gm0 * ldc + gn0;

    const int  t8 = (tid & 7) * 8;
    const int  to = tid >> 3;
    const long stepA = AT ? 64 * lda : 64;
    const long stepB = BT ? 64 : 64 * ldb;
    const float*  pAf = Af + offA + (long)to * lda + t8;
    const __bf16* pAb = Ab + offA + (long)to * lda + t8;
    const float*  pBf = Bf + offB + (long)to * ldb + t8;

    const int  half = lane >> 4;
    const int  l15  = lane & 15;
    const __bf16* arow = &As[tm * 16 + l15][0];
    const __bf16* brow = &Bs[tn * 16 + l15][0];

    v8f acc = {};

    for (int k0 = 0; k0 < K; k0 += 64) {
        v4f  af0 = {}, af1 = {};
        v8bf a8  = {};
        if constexpr (ABF) a8 = *(const v8bf*)pAb;
        else { af0 = *(const v4f*)pAf; af1 = *(const v4f*)(pAf + 4); }
        const v4f bf0 = *(const v4f*)pBf;
        const v4f bf1 = *(const v4f*)(pBf + 4);
        if (k0 + 64 < K) {
            __builtin_prefetch(ABF ? (const void*)(pAb + stepA)
                                   : (const void*)(pAf + stepA), 0, 3);
            __builtin_prefetch(pBf + stepB, 0, 3);
        }

        v8bf as;
        if constexpr (ABF) as = a8; else as = pack8(af0, af1);
        const v8bf bs = pack8(bf0, bf1);

        if constexpr (AT) {
            #pragma unroll
            for (int j = 0; j < 8; ++j) As[t8 + j][to] = as[j];
        } else *(v8bf*)&As[to][t8] = as;
        if constexpr (BT) *(v8bf*)&Bs[to][t8] = bs;
        else {
            #pragma unroll
            for (int j = 0; j < 8; ++j) Bs[t8 + j][to] = bs[j];
        }
        pAf += stepA; pAb += stepA; pBf += stepB;
        __syncthreads();

        #pragma unroll
        for (int kb = 0; kb < 64; kb += 32) {
            const v16bf a = ldfragA(arow, kb, half);
            const v16bf b = ldfragB(brow, kb, half);
            acc = __builtin_amdgcn_wmma_f32_16x16x32_bf16(
                false, a, false, b, (short)0, acc, false, false);
        }
        __syncthreads();
    }

    const int col = tn * 16 + l15;
    float badd = 0.0f;
    if (BIAS) badd = bias[zh * sBiasH + gn0 + col];
    #pragma unroll
    for (int r = 0; r < 8; ++r) {
        const int  row = tm * 16 + half * 8 + r;
        const float v  = acc[r] + badd;
        const long ci  = offC + (long)row * ldc + col;
        if (OBF) ((__bf16*)Cv)[ci] = (__bf16)v;
        else     ((float*)Cv)[ci]  = v;
    }
}

// ---------------------------------------------------------------------------
__global__ __launch_bounds__(256)
void mean_rows(const float* __restrict__ x, float* __restrict__ xbar)
{
    const long CE = 2048L * 512;
    const long i  = (long)blockIdx.x * blockDim.x + threadIdx.x;
    if (i >= CE) return;
    float s = 0.0f;
    #pragma unroll 8
    for (int b = 0; b < 64; ++b) s += x[(long)b * CE + i];
    xbar[i] = s * (1.0f / 64.0f);
}

// softmax over D=64 within each head, times D^-0.5 = 0.125.  One wave per row.
__global__ __launch_bounds__(256)
void softmax_q(float* __restrict__ q)
{
    const int lane = threadIdx.x & 31;
    const int wave = threadIdx.x >> 5;
    const int row  = blockIdx.x * 8 + wave;
    const int c    = row >> 3, h = row & 7;
    const long base = (long)c * 512 + h * 64;
    float v0 = q[base + lane], v1 = q[base + 32 + lane];
    float m = fmaxf(v0, v1);
    for (int o = 16; o; o >>= 1) m = fmaxf(m, __shfl_xor(m, o));
    float e0 = __expf(v0 - m), e1 = __expf(v1 - m);
    float s = e0 + e1;
    for (int o = 16; o; o >>= 1) s += __shfl_xor(s, o);
    const float inv = 0.125f / s;
    q[base + lane]      = e0 * inv;
    q[base + 32 + lane] = e1 * inv;
}

// softmax over C=2048 per (h,d) column (512 columns).  One block per column.
__global__ __launch_bounds__(256)
void softmax_k(float* __restrict__ kk)
{
    __shared__ float red[256];
    const int col = blockIdx.x;
    const int t   = threadIdx.x;
    float v[8];
    float m = -3.4e38f;
    #pragma unroll
    for (int j = 0; j < 8; ++j) {
        v[j] = kk[(long)(j * 256 + t) * 512 + col];
        m = fmaxf(m, v[j]);
    }
    red[t] = m; __syncthreads();
    for (int s = 128; s; s >>= 1) { if (t < s) red[t] = fmaxf(red[t], red[t + s]); __syncthreads(); }
    m = red[0]; __syncthreads();
    float sum = 0.0f;
    #pragma unroll
    for (int j = 0; j < 8; ++j) { v[j] = __expf(v[j] - m); sum += v[j]; }
    red[t] = sum; __syncthreads();
    for (int s = 128; s; s >>= 1) { if (t < s) red[t] += red[t + s]; __syncthreads(); }
    const float inv = 1.0f / red[0];
    #pragma unroll
    for (int j = 0; j < 8; ++j) kk[(long)(j * 256 + t) * 512 + col] = v[j] * inv;
}

// ---------------------------------------------------------------------------
extern "C" void kernel_launch(void* const* d_in, const int* in_sizes, int n_in,
                              void* d_out, int out_size, void* d_ws, size_t ws_size,
                              hipStream_t stream)
{
    const float* x  = (const float*)d_in[0];   // [64,2048,512]
    const float* Wq = (const float*)d_in[1];
    const float* bq = (const float*)d_in[2];
    const float* Wk = (const float*)d_in[3];
    const float* bk = (const float*)d_in[4];
    const float* Wv = (const float*)d_in[5];
    const float* bv = (const float*)d_in[6];

    float* out  = (float*)d_out;               // [64,2048,512]
    float* attn = out + 64L * 2048 * 512;      // [8,2048,2048]

    char*   ws   = (char*)d_ws;
    float*  xbar = (float*)(ws);               // [2048,512]        4 MB
    float*  qm   = (float*)(ws + (4L  << 20)); // q proj / qs       4 MB
    float*  km   = (float*)(ws + (8L  << 20)); // k proj / ks       4 MB
    float*  ctx  = (float*)(ws + (12L << 20)); // [64,8,64,64]      8 MB
    __bf16* kx   = (__bf16*)(ws + (20L << 20));// [64,512,512] bf16 32 MB
    (void)in_sizes; (void)n_in; (void)out_size; (void)ws_size;

    // 1. xbar = mean_R(x)  (mean commutes with the q/k Linear layers)
    mean_rows<<<4096, 256, 0, stream>>>(x, xbar);

    // 2. q/k projections of the mean:  qm = xbar@Wq^T+bq ; km = xbar@Wk^T+bk
    gemm128<false, false, true, true, false><<<dim3(4, 16, 1), 512, 0, stream>>>(
        xbar, 512, Wq, 512, bq, qm, 512, /*K*/512, /*ZH*/1,
        0, 0,  0, 0,  0, 0,  0);
    gemm128<false, false, true, true, false><<<dim3(4, 16, 1), 512, 0, stream>>>(
        xbar, 512, Wk, 512, bk, km, 512, 512, 1,
        0, 0,  0, 0,  0, 0,  0);

    // 3. softmaxes (qs over D, scaled; ks over C)
    softmax_q<<<2048, 256, 0, stream>>>(qm);
    softmax_k<<<512, 256, 0, stream>>>(km);

    // 4. kx[b] = ks^T @ x[b]   (M=512 hd, N=512 e, K=2048) -> bf16
    gemm128<true, false, false, false, true><<<dim3(4, 4, 64), 512, 0, stream>>>(
        km, 512, x, 512, nullptr, kx, 512, /*K*/2048, /*ZH*/1,
        0, 0,  2048L * 512, 0,  512L * 512, 0,  0);

    // 5. ctx[b,h] = kx[b, h*64: , :] @ Wv_h^T + bv_h   (64x64, K=512)
    gemm64x64<false, true, true, true, false><<<dim3(1, 1, 512), 512, 0, stream>>>(
        kx, 512, Wv, 512, bv, ctx, 64, /*K*/512, /*ZH*/8,
        /*sA*/ 512L * 512, 64L * 512,  /*sB*/ 0, 64L * 512,
        /*sC*/ 8L * 4096, 4096,  /*bias*/ 64);

    // 6. out[b,:,h*64:] = qs_h @ ctx[b,h]   (M=2048, N=64, K=64)
    gemm64x64<false, false, false, false, false><<<dim3(1, 32, 512), 512, 0, stream>>>(
        qm, 512, (const float*)ctx, 64, nullptr, out, 512, /*K*/64, /*ZH*/8,
        /*sA*/ 0, 64,  /*sB*/ 8L * 4096, 4096,
        /*sC*/ 2048L * 512, 64,  0);

    // 7. attn[h] = qs_h @ ks_h^T   (M=2048, N=2048, K=64)
    gemm128<false, false, true, false, false><<<dim3(16, 16, 8), 512, 0, stream>>>(
        qm, 512, km, 512, nullptr, attn, 2048, /*K*/64, /*ZH*/8,
        /*sA*/ 0, 64,  /*sB*/ 0, 64,
        /*sC*/ 0, 2048L * 2048,  0);
}
